// SelfSwinTransformerBlock_69750268887304
// MI455X (gfx1250) — matrix operand order repaired and checked by reading, not compiled, not run on hardware
//
#include <hip/hip_runtime.h>
#include <hip/hip_bf16.h>
#include <math.h>
#include <stdint.h>

typedef __attribute__((ext_vector_type(16))) _Float16 v16h;
typedef __attribute__((ext_vector_type(8)))  _Float16 v8h;
typedef __attribute__((ext_vector_type(8)))  float    v8f;

#define RES_D   28
#define C_DIM   192
#define NHEAD   6
#define HD      32
#define NTOK    343
#define NPAD    352
#define NWIN    64      // windows per batch (4x4x4)
#define BWIN    128     // total windows (B=2)
#define LTOK    21952
#define TOTTOK  43904
#define SHIFT   3

// ---------------- WMMA fragment helpers (ISA 7.12.2 layouts) ----------------

__device__ __forceinline__ v16h load_fragA(const _Float16* base, int ld, int lane, int k0) {
  // A 16x32 f16: row = lane%16; K = (e/8)*16 + (lane/16)*8 + (e%8)
  int row = lane & 15, g = lane >> 4;
  const _Float16* p = base + row * ld + k0 + g * 8;
  v8h lo = *(const v8h*)(p);
  v8h hi = *(const v8h*)(p + 16);
  return __builtin_shufflevector(lo, hi, 0,1,2,3,4,5,6,7,8,9,10,11,12,13,14,15);
}

__device__ __forceinline__ v16h load_fragB(const _Float16* base, int ld, int lane, int n0, int k0) {
  // B 32x16 f16: col = lane%16; K = (e/8)*16 + (lane/16)*8 + (e%8)
  int col = n0 + (lane & 15), g = lane >> 4;
  const _Float16* p = base + (size_t)col * ld + k0 + g * 8;
  v8h lo = *(const v8h*)(p);
  v8h hi = *(const v8h*)(p + 16);
  return __builtin_shufflevector(lo, hi, 0,1,2,3,4,5,6,7,8,9,10,11,12,13,14,15);
}

__device__ __forceinline__ v8f wmma16(v16h a, v16h b, v8f c) {
  return __builtin_amdgcn_wmma_f32_16x16x32_f16(false, a, false, b, (short)0, c, false, false);
}

__device__ __forceinline__ int swregion(int c) { return c < 21 ? 0 : (c < 25 ? 1 : 2); }

// Direct global->LDS async copy (CDNA5 path, ASYNCcnt-tracked).
// halves must be a multiple of 8 (16-byte granules); both pointers 16B-aligned.
__device__ __forceinline__ void async_copy_to_lds(_Float16* dst, const _Float16* src,
                                                  int halves, int tid, int nthreads) {
  int n16 = halves >> 3;
  for (int i = tid; i < n16; i += nthreads) {
    unsigned ldsoff = (unsigned)(uintptr_t)(dst + i * 8);   // LDS offset = low 32 bits
    const _Float16* g = src + i * 8;
    asm volatile("global_load_async_to_lds_b128 %0, %1, off"
                 :: "v"(ldsoff), "v"(g) : "memory");
  }
  asm volatile("s_wait_asynccnt 0x0" ::: "memory");
}

// ---------------- tiny kernels ----------------

__global__ void k_w2h(const float* __restrict__ src, _Float16* __restrict__ dst, int n) {
  int i = blockIdx.x * blockDim.x + threadIdx.x;
  if (i < n) dst[i] = (_Float16)src[i];
}

__device__ __forceinline__ float cpb_coord(int a) {
  float r = (float)(a - 6) * (8.0f / 6.0f);
  float s = (r > 0.f) ? 1.f : ((r < 0.f) ? -1.f : 0.f);
  return s * log2f(fabsf(r) + 1.f) * (1.f / 3.f);   // /log2(8)
}

// hidden MLP of the continuous position bias table: hb[2197][6]
__global__ void k_cpb(const float* __restrict__ w1, const float* __restrict__ b1,
                      const float* __restrict__ w2, float* __restrict__ hb) {
  __shared__ float acc[NHEAD];
  int ti = blockIdx.x, j = threadIdx.x;
  if (j < NHEAD) acc[j] = 0.f;
  __syncthreads();
  int a = ti / 169, b = (ti / 13) % 13, c = ti % 13;
  float t0 = cpb_coord(a), t1 = cpb_coord(b), t2 = cpb_coord(c);
  float h = t0 * w1[j*3] + t1 * w1[j*3+1] + t2 * w1[j*3+2] + b1[j];
  h = fmaxf(h, 0.f);
  #pragma unroll
  for (int o = 0; o < NHEAD; ++o) atomicAdd(&acc[o], h * w2[o*512 + j]);
  __syncthreads();
  if (j < NHEAD) hb[ti*NHEAD + j] = acc[j];
}

// shift-roll + window partition, f32 -> f16, zero-padded to 352 tokens
__global__ void k_part(const float* __restrict__ x, _Float16* __restrict__ xw) {
  int w = blockIdx.x / NPAD;
  int t = blockIdx.x % NPAD;
  int c = threadIdx.x;
  float v = 0.f;
  if (t < NTOK) {
    int b = w / NWIN, wl = w % NWIN;
    int wx = wl / 16, wy = (wl / 4) % 4, wz = wl % 4;
    int t0 = t / 49, t1 = (t / 7) % 7, t2 = t % 7;
    int hs = (wx*7 + t0 + SHIFT) % RES_D;
    int ws = (wy*7 + t1 + SHIFT) % RES_D;
    int ds = (wz*7 + t2 + SHIFT) % RES_D;
    v = x[((size_t)b * LTOK + (hs*RES_D + ws)*RES_D + ds) * C_DIM + c];
  }
  xw[((size_t)w * NPAD + t) * C_DIM + c] = (_Float16)v;
}

// ---------------- QKV GEMM: [352 x 192] @ [192 x 576] per window ----------------
__global__ void k_qkv(const _Float16* __restrict__ xw, const _Float16* __restrict__ wq,
                      const float* __restrict__ qb, const float* __restrict__ vb,
                      _Float16* __restrict__ qkv) {
  __shared__ __align__(32) _Float16 As[16 * C_DIM];
  int win = blockIdx.x, m0 = blockIdx.y * 16;
  const _Float16* src = xw + ((size_t)win * NPAD + m0) * C_DIM;
  async_copy_to_lds(As, src, 16 * C_DIM, threadIdx.x, blockDim.x);
  __syncthreads();
  int wave = threadIdx.x >> 5, lane = threadIdx.x & 31;
  for (int nt = wave; nt < 36; nt += 8) {
    v16h bfr[6];
    #pragma unroll
    for (int ks = 0; ks < 6; ++ks) bfr[ks] = load_fragB(wq, C_DIM, lane, nt*16, ks*32);
    v8f acc = {};
    #pragma unroll
    for (int ks = 0; ks < 6; ++ks)
      acc = wmma16(load_fragA(As, C_DIM, lane, ks*32), bfr[ks], acc);
    int nl = lane & 15, g = lane >> 4;
    int ncol = nt * 16 + nl;
    float bias = (ncol < 192) ? qb[ncol] : ((ncol < 384) ? 0.f : vb[ncol - 384]);
    #pragma unroll
    for (int r = 0; r < 8; ++r) {
      int M = r + 8 * g;
      qkv[((size_t)win * NPAD + m0 + M) * 576 + ncol] = (_Float16)(acc[r] + bias);
    }
  }
}

// ---------------- q/k L2 normalize + V transpose, one wave per vector ----------------
__global__ void k_norm(const _Float16* __restrict__ qkv, _Float16* __restrict__ qn,
                       _Float16* __restrict__ kn, _Float16* __restrict__ vT) {
  int gid = blockIdx.x * 8 + (threadIdx.x >> 5);
  int lane = threadIdx.x & 31;
  if (gid >= BWIN * NHEAD * NPAD) return;
  int t = gid % NPAD;
  int wh = gid / NPAD;
  int head = wh % NHEAD, win = wh / NHEAD;
  const _Float16* base = qkv + ((size_t)win * NPAD + t) * 576;
  float q = (float)base[head*HD + lane];
  float k = (float)base[192 + head*HD + lane];
  float v = (float)base[384 + head*HD + lane];
  float qs = q * q, ks = k * k;
  #pragma unroll
  for (int o = 16; o > 0; o >>= 1) { qs += __shfl_xor(qs, o, 32); ks += __shfl_xor(ks, o, 32); }
  float qo = q / fmaxf(sqrtf(qs), 1e-12f);
  float ko = k / fmaxf(sqrtf(ks), 1e-12f);
  size_t hoff = (size_t)win * NHEAD + head;
  qn[(hoff * NPAD + t) * HD + lane] = (_Float16)qo;
  kn[(hoff * NPAD + t) * HD + lane] = (_Float16)ko;
  vT[(hoff * HD + lane) * NPAD + t] = (_Float16)v;
}

// ---------------- attention per (window,head, 16-row tile) ----------------
__global__ void k_attn(const _Float16* __restrict__ qn, const _Float16* __restrict__ kn,
                       const _Float16* __restrict__ vT, const float* __restrict__ hb,
                       const float* __restrict__ logit_scale, _Float16* __restrict__ attn_out) {
  __shared__ __align__(32) float    Ssc[16 * NPAD];
  __shared__ __align__(32) _Float16 Pp[16 * NPAD];
  __shared__ __align__(32) float    Part[8 * 16 * 16];
  int wh = blockIdx.x;                 // win*NHEAD + head
  int head = wh % NHEAD, win = wh / NHEAD;
  int m0 = blockIdx.y * 16;
  int wave = threadIdx.x >> 5, lane = threadIdx.x & 31;
  int wl = win % NWIN;
  int wx = wl / 16, wy = (wl / 4) % 4, wz = wl % 4;
  size_t hoff = (size_t)wh;
  float scale = expf(fminf(logit_scale[head], 4.6051702f));   // log(100)
  const _Float16* qbase = qn + hoff * NPAD * HD;
  const _Float16* kbase = kn + hoff * NPAD * HD;

  // ---- phase 1: S = (qn kn^T)*scale + rpb + mask ----
  v16h afrag = load_fragA(qbase + (size_t)m0 * HD, HD, lane, 0);
  for (int nt = wave; nt < 22; nt += 8) {
    v8f acc = {};
    acc = wmma16(afrag, load_fragB(kbase, HD, lane, nt*16, 0), acc);
    int nl = lane & 15, g = lane >> 4;
    int j = nt * 16 + nl;
    int jj = j > 342 ? 342 : j;
    int j0 = jj / 49, j1 = (jj / 7) % 7, j2 = jj % 7;
    int lj = swregion(wx*7 + j0)*9 + swregion(wy*7 + j1)*3 + swregion(wz*7 + j2);
    #pragma unroll
    for (int r = 0; r < 8; ++r) {
      int M = r + 8 * g, i = m0 + M;
      float s;
      if (j >= NTOK) s = -1e30f;
      else {
        int ii = i > 342 ? 342 : i;
        int i0 = ii / 49, i1 = (ii / 7) % 7, i2 = ii % 7;
        int idx = ((i0 - j0 + 6) * 13 + (i1 - j1 + 6)) * 13 + (i2 - j2 + 6);
        float rpb = 16.f / (1.f + expf(-hb[idx * NHEAD + head]));
        int li = swregion(wx*7 + i0)*9 + swregion(wy*7 + i1)*3 + swregion(wz*7 + i2);
        s = acc[r] * scale + rpb + ((li == lj) ? 0.f : -100.f);
      }
      Ssc[M * NPAD + j] = s;
    }
  }
  __syncthreads();

  // ---- phase 2: softmax, 2 rows per wave ----
  #pragma unroll
  for (int rr = 0; rr < 2; ++rr) {
    int row = wave * 2 + rr;
    float mx = -1e30f;
    for (int j = lane; j < NPAD; j += 32) mx = fmaxf(mx, Ssc[row*NPAD + j]);
    #pragma unroll
    for (int o = 16; o > 0; o >>= 1) mx = fmaxf(mx, __shfl_xor(mx, o, 32));
    float sum = 0.f;
    for (int j = lane; j < NPAD; j += 32) sum += expf(Ssc[row*NPAD + j] - mx);
    #pragma unroll
    for (int o = 16; o > 0; o >>= 1) sum += __shfl_xor(sum, o, 32);
    float inv = 1.f / sum;
    for (int j = lane; j < NPAD; j += 32)
      Pp[row*NPAD + j] = (_Float16)(expf(Ssc[row*NPAD + j] - mx) * inv);
  }
  __syncthreads();

  // ---- phase 3: O = P V  (4-way K-split per 16-col group, LDS reduce) ----
  int grp = wave >> 2, kidx = wave & 3;
  const _Float16* vbase = vT + hoff * HD * NPAD;
  v8f acc = {};
  for (int ks = kidx; ks < 11; ks += 4)
    acc = wmma16(load_fragA(Pp, NPAD, lane, ks*32),
                 load_fragB(vbase, NPAD, lane, grp*16, ks*32), acc);
  {
    int nl = lane & 15, g = lane >> 4;
    #pragma unroll
    for (int r = 0; r < 8; ++r) Part[(wave * 16 + (r + 8*g)) * 16 + nl] = acc[r];
  }
  __syncthreads();
  for (int e = threadIdx.x; e < 512; e += blockDim.x) {
    int g2 = e >> 8, M = (e >> 4) & 15, nl = e & 15;
    float s = 0.f;
    #pragma unroll
    for (int w2 = 0; w2 < 4; ++w2) s += Part[((g2*4 + w2) * 16 + M) * 16 + nl];
    attn_out[((size_t)win * NPAD + m0 + M) * C_DIM + head*HD + g2*16 + nl] = (_Float16)s;
  }
}

// ---------------- proj GEMM + window reverse + un-roll scatter ----------------
__global__ void k_proj(const _Float16* __restrict__ attn, const _Float16* __restrict__ wp,
                       const float* __restrict__ pb, float* __restrict__ xa) {
  __shared__ __align__(32) _Float16 As[16 * C_DIM];
  int win = blockIdx.x, m0 = blockIdx.y * 16;
  const _Float16* src = attn + ((size_t)win * NPAD + m0) * C_DIM;
  async_copy_to_lds(As, src, 16 * C_DIM, threadIdx.x, blockDim.x);
  __syncthreads();
  int wave = threadIdx.x >> 5, lane = threadIdx.x & 31;
  int b = win / NWIN, wl = win % NWIN;
  int wx = wl / 16, wy = (wl / 4) % 4, wz = wl % 4;
  for (int nt = wave; nt < 12; nt += 8) {
    v16h bfr[6];
    #pragma unroll
    for (int ks = 0; ks < 6; ++ks) bfr[ks] = load_fragB(wp, C_DIM, lane, nt*16, ks*32);
    v8f acc = {};
    #pragma unroll
    for (int ks = 0; ks < 6; ++ks)
      acc = wmma16(load_fragA(As, C_DIM, lane, ks*32), bfr[ks], acc);
    int nl = lane & 15, g = lane >> 4;
    int ncol = nt * 16 + nl;
    float bias = pb[ncol];
    #pragma unroll
    for (int r = 0; r < 8; ++r) {
      int t = m0 + r + 8 * g;
      if (t >= NTOK) continue;
      int t0 = t / 49, t1 = (t / 7) % 7, t2 = t % 7;
      int hu = (wx*7 + t0 + SHIFT) % RES_D;
      int wu = (wy*7 + t1 + SHIFT) % RES_D;
      int du = (wz*7 + t2 + SHIFT) % RES_D;
      xa[((size_t)b * LTOK + (hu*RES_D + wu)*RES_D + du) * C_DIM + ncol] = acc[r] + bias;
    }
  }
}

// ---------------- LN + residual (one token per block) ----------------
__global__ void k_ln_res(const float* __restrict__ xin, const float* __restrict__ y,
                         const float* __restrict__ g, const float* __restrict__ bb,
                         float* __restrict__ out, _Float16* __restrict__ outh) {
  __shared__ float s1[256], s2[256];
  int row = blockIdx.x, tid = threadIdx.x;
  float v = (tid < C_DIM) ? y[(size_t)row * C_DIM + tid] : 0.f;
  s1[tid] = v; s2[tid] = v * v;
  __syncthreads();
  #pragma unroll
  for (int o = 128; o > 0; o >>= 1) {
    if (tid < o) { s1[tid] += s1[tid + o]; s2[tid] += s2[tid + o]; }
    __syncthreads();
  }
  float mean = s1[0] * (1.f / C_DIM);
  float var  = s2[0] * (1.f / C_DIM) - mean * mean;
  float inv  = rsqrtf(var + 1e-5f);
  if (tid < C_DIM) {
    float o = xin[(size_t)row * C_DIM + tid] + (v - mean) * inv * g[tid] + bb[tid];
    out[(size_t)row * C_DIM + tid] = o;
    if (outh) outh[(size_t)row * C_DIM + tid] = (_Float16)o;
  }
}

// ---------------- fc1 GEMM + exact GELU ----------------
__global__ void k_fc1(const _Float16* __restrict__ x1h, const _Float16* __restrict__ w,
                      const float* __restrict__ bias, _Float16* __restrict__ hh) {
  __shared__ __align__(32) _Float16 As[16 * C_DIM];
  int m0 = blockIdx.x * 16;
  async_copy_to_lds(As, x1h + (size_t)m0 * C_DIM, 16 * C_DIM, threadIdx.x, blockDim.x);
  __syncthreads();
  int wave = threadIdx.x >> 5, lane = threadIdx.x & 31;
  for (int nt = wave; nt < 48; nt += 8) {
    v16h bfr[6];
    #pragma unroll
    for (int ks = 0; ks < 6; ++ks) bfr[ks] = load_fragB(w, C_DIM, lane, nt*16, ks*32);
    v8f acc = {};
    #pragma unroll
    for (int ks = 0; ks < 6; ++ks)
      acc = wmma16(load_fragA(As, C_DIM, lane, ks*32), bfr[ks], acc);
    int nl = lane & 15, g = lane >> 4;
    int ncol = nt * 16 + nl;
    float bv = bias[ncol];
    #pragma unroll
    for (int r = 0; r < 8; ++r) {
      float vv = acc[r] + bv;
      vv = 0.5f * vv * (1.f + erff(vv * 0.70710678f));
      hh[((size_t)m0 + r + 8*g) * 768 + ncol] = (_Float16)vv;
    }
  }
}

// ---------------- fc2 GEMM ----------------
__global__ void k_fc2(const _Float16* __restrict__ hh, const _Float16* __restrict__ w,
                      const float* __restrict__ bias, float* __restrict__ m) {
  __shared__ __align__(32) _Float16 As[16 * 768];
  int m0 = blockIdx.x * 16;
  async_copy_to_lds(As, hh + (size_t)m0 * 768, 16 * 768, threadIdx.x, blockDim.x);
  __syncthreads();
  int wave = threadIdx.x >> 5, lane = threadIdx.x & 31;
  for (int nt = wave; nt < 12; nt += 8) {
    v8f acc = {};
    #pragma unroll
    for (int kb = 0; kb < 4; ++kb) {      // 24 k-steps in 4 chunks of 6
      v16h bfr[6];
      #pragma unroll
      for (int ks = 0; ks < 6; ++ks)
        bfr[ks] = load_fragB(w, 768, lane, nt*16, (kb*6 + ks)*32);
      #pragma unroll
      for (int ks = 0; ks < 6; ++ks)
        acc = wmma16(load_fragA(As, 768, lane, (kb*6 + ks)*32), bfr[ks], acc);
    }
    int nl = lane & 15, g = lane >> 4;
    int ncol = nt * 16 + nl;
    float bv = bias[ncol];
    #pragma unroll
    for (int r = 0; r < 8; ++r)
      m[((size_t)m0 + r + 8*g) * C_DIM + ncol] = acc[r] + bv;
  }
}

// ---------------- driver ----------------
extern "C" void kernel_launch(void* const* d_in, const int* in_sizes, int n_in,
                              void* d_out, int out_size, void* d_ws, size_t ws_size,
                              hipStream_t stream) {
  (void)in_sizes; (void)n_in; (void)out_size; (void)ws_size;
  const float* x        = (const float*)d_in[0];
  const float* qkv_w    = (const float*)d_in[1];
  const float* q_bias   = (const float*)d_in[2];
  const float* v_bias   = (const float*)d_in[3];
  const float* lscale   = (const float*)d_in[4];
  const float* cpb_w1   = (const float*)d_in[5];
  const float* cpb_b1   = (const float*)d_in[6];
  const float* cpb_w2   = (const float*)d_in[7];
  const float* proj_w   = (const float*)d_in[8];
  const float* proj_b   = (const float*)d_in[9];
  const float* norm1_g  = (const float*)d_in[10];
  const float* norm1_b  = (const float*)d_in[11];
  const float* fc1_w    = (const float*)d_in[12];
  const float* fc1_b    = (const float*)d_in[13];
  const float* fc2_w    = (const float*)d_in[14];
  const float* fc2_b    = (const float*)d_in[15];
  const float* norm2_g  = (const float*)d_in[16];
  const float* norm2_b  = (const float*)d_in[17];
  float* out = (float*)d_out;

  char* p = (char*)d_ws;
  auto alloc = [&](size_t bytes) -> void* {
    void* r = (void*)p;
    p += (bytes + 255) & ~(size_t)255;
    return r;
  };
  float*     hb      = (float*)    alloc((size_t)2197 * NHEAD * 4);
  _Float16*  wqkv_h  = (_Float16*) alloc((size_t)576 * 192 * 2);
  _Float16*  wproj_h = (_Float16*) alloc((size_t)192 * 192 * 2);
  _Float16*  wfc1_h  = (_Float16*) alloc((size_t)768 * 192 * 2);
  _Float16*  wfc2_h  = (_Float16*) alloc((size_t)192 * 768 * 2);
  _Float16*  xw_h    = (_Float16*) alloc((size_t)BWIN * NPAD * C_DIM * 2);
  _Float16*  qkv_h   = (_Float16*) alloc((size_t)BWIN * NPAD * 576 * 2);
  _Float16*  qn_h    = (_Float16*) alloc((size_t)BWIN * NHEAD * NPAD * HD * 2);
  _Float16*  kn_h    = (_Float16*) alloc((size_t)BWIN * NHEAD * NPAD * HD * 2);
  _Float16*  vT_h    = (_Float16*) alloc((size_t)BWIN * NHEAD * HD * NPAD * 2);
  _Float16*  attn_h  = (_Float16*) alloc((size_t)BWIN * NPAD * C_DIM * 2);
  float*     xa      = (float*)    alloc((size_t)TOTTOK * C_DIM * 4);
  float*     x1      = (float*)    alloc((size_t)TOTTOK * C_DIM * 4);
  _Float16*  x1h     = (_Float16*) alloc((size_t)TOTTOK * C_DIM * 2);
  _Float16*  hh      = (_Float16*) alloc((size_t)TOTTOK * 768 * 2);
  float*     mbuf    = (float*)    alloc((size_t)TOTTOK * C_DIM * 4);

  // weight conversions
  k_w2h<<<(576*192 + 255)/256, 256, 0, stream>>>(qkv_w, wqkv_h, 576*192);
  k_w2h<<<(192*192 + 255)/256, 256, 0, stream>>>(proj_w, wproj_h, 192*192);
  k_w2h<<<(768*192 + 255)/256, 256, 0, stream>>>(fc1_w, wfc1_h, 768*192);
  k_w2h<<<(192*768 + 255)/256, 256, 0, stream>>>(fc2_w, wfc2_h, 192*768);

  // CPB table MLP
  k_cpb<<<2197, 512, 0, stream>>>(cpb_w1, cpb_b1, cpb_w2, hb);

  // shift + window partition
  k_part<<<BWIN * NPAD, C_DIM, 0, stream>>>(x, xw_h);

  // QKV GEMM
  k_qkv<<<dim3(BWIN, NPAD/16), 256, 0, stream>>>(xw_h, wqkv_h, q_bias, v_bias, qkv_h);

  // normalize q/k, transpose v
  k_norm<<<(BWIN * NHEAD * NPAD + 7) / 8, 256, 0, stream>>>(qkv_h, qn_h, kn_h, vT_h);

  // attention
  k_attn<<<dim3(BWIN * NHEAD, NPAD/16), 256, 0, stream>>>(qn_h, kn_h, vT_h, hb, lscale, attn_h);

  // proj + window reverse + un-roll
  k_proj<<<dim3(BWIN, NPAD/16), 256, 0, stream>>>(attn_h, wproj_h, proj_b, xa);

  // x1 = x + LN1(xa)
  k_ln_res<<<TOTTOK, 256, 0, stream>>>(x, xa, norm1_g, norm1_b, x1, x1h);

  // MLP
  k_fc1<<<TOTTOK/16, 256, 0, stream>>>(x1h, wfc1_h, fc1_b, hh);
  k_fc2<<<TOTTOK/16, 256, 0, stream>>>(hh, wfc2_h, fc2_b, mbuf);

  // out = x1 + LN2(m)
  k_ln_res<<<TOTTOK, 256, 0, stream>>>(x1, mbuf, norm2_g, norm2_b, out, (_Float16*)nullptr);
}
